// GraphEncoder_20298015441381
// MI455X (gfx1250) — matrix operand order repaired and checked by reading, not compiled, not run on hardware
//
#include <hip/hip_runtime.h>
#include <hip/hip_bf16.h>

#define NN 100000
#define NE 1600000
#define IN_NODE 64
#define IN_EDGE 16
#define HID 128
#define OUT_DIM 128
#define N_CONVS 3
#define NEG_SLOPE 0.01f
#define LN_EPS 1e-5f

typedef __attribute__((ext_vector_type(16))) __bf16 v16bf;
typedef __attribute__((ext_vector_type(8)))  __bf16 v8bf;
typedef __attribute__((ext_vector_type(8)))  float  v8f;
typedef __attribute__((ext_vector_type(4)))  float  v4f;

#define TILES_PER_BLOCK 4

// ---------------------------------------------------------------------------
// Core 16x16 WMMA tile: A = 16 x K (row-major bf16 in LDS),
// wT = 128 x K (column-of-W major bf16 in LDS, i.e. W transposed).
// Returns the 16x16 f32 accumulator for output columns [16*wave, 16*wave+16).
// Operand packing per CDNA5 ISA 7.12.2 (wave32):
//   A bf16 16x32: lanes 0-15 -> K {0..7, 16..23}, lanes 16-31 -> K {8..15, 24..31}
//   B bf16 32x16: lanes 0-15 -> K {0..7, 16..23}, lanes 16-31 -> K {8..15, 24..31} (N = lane&15)
// ---------------------------------------------------------------------------
template<int K>
__device__ __forceinline__ v8f wmma_tile(const __bf16* aT, const __bf16* wT,
                                         int lane, int wave)
{
    const int row = lane & 15;
    const int hi  = lane >> 4;           // 0 or 1
    const int gcol = wave * 16 + row;    // global output column for B operand
    v8f c = {};
#pragma unroll
    for (int kb = 0; kb < K / 32; ++kb) {
        union { v16bf v; v8bf h[2]; } a, b;
        const __bf16* ap = aT + row * K + kb * 32 + hi * 8;
        a.h[0] = *(const v8bf*)(ap);          // K = base+0..7
        a.h[1] = *(const v8bf*)(ap + 16);     // K = base+16..23
        const __bf16* bp = wT + gcol * K + kb * 32 + hi * 8;
        b.h[0] = *(const v8bf*)(bp);          // K = base+0..7
        b.h[1] = *(const v8bf*)(bp + 16);     // K = base+16..23
        c = __builtin_amdgcn_wmma_f32_16x16x32_bf16(false, a.v, false, b.v,
                                                    (short)0, c, false, false);
    }
    return c;
}

// ---------------------------------------------------------------------------
// f32 -> bf16 weight conversion
// ---------------------------------------------------------------------------
__global__ void __launch_bounds__(256)
cvt_bf16_kernel(const float* __restrict__ in, __bf16* __restrict__ out, int n)
{
    int i = blockIdx.x * 256 + threadIdx.x;
    if (i < n) out[i] = (__bf16)in[i];
}

// ---------------------------------------------------------------------------
// zero f32 buffer (vectorized)
// ---------------------------------------------------------------------------
__global__ void __launch_bounds__(256)
zero_kernel(float* __restrict__ p, long long n4)
{
    long long i = (long long)blockIdx.x * 256 + threadIdx.x;
    if (i < n4) ((v4f*)p)[i] = (v4f){0.f, 0.f, 0.f, 0.f};
}

// ---------------------------------------------------------------------------
// h = x @ W_node + b_node ; skip = h     (M = NN, K = 64, N = 128)
// ---------------------------------------------------------------------------
__global__ void __launch_bounds__(256)
node_kernel(const float* __restrict__ x, const __bf16* __restrict__ Wn,
            const float* __restrict__ bias,
            float* __restrict__ h, float* __restrict__ skip, int nTiles)
{
    constexpr int K = IN_NODE;
    __shared__ __align__(16) __bf16 wT[HID * K];   // 16 KB, W transposed
    __shared__ __align__(16) __bf16 aT[16 * K];    // 2 KB
    __shared__ float bS[HID];

    const int tid = threadIdx.x;
    for (int i = tid; i < K * HID; i += 256) {     // Wn is [K][128] row-major
        int k = i >> 7, col = i & 127;
        wT[col * K + k] = Wn[i];
    }
    if (tid < HID) bS[tid] = bias[tid];

    const int lane = tid & 31, wave = tid >> 5;
    const int tile0 = blockIdx.x * TILES_PER_BLOCK;
    for (int t = 0; t < TILES_PER_BLOCK; ++t) {
        int tile = tile0 + t;
        if (tile >= nTiles) break;                 // uniform across block
        long long m0 = (long long)tile * 16;
        __syncthreads();
        for (int i = tid; i < 16 * K; i += 256) {
            int r = i / K, k = i % K;
            aT[i] = (__bf16)x[(m0 + r) * K + k];
        }
        __syncthreads();
        v8f c = wmma_tile<K>(aT, wT, lane, wave);
        const int row = lane & 15, hi = lane >> 4;
        const int gcol = wave * 16 + row;
        const float bv = bS[gcol];
#pragma unroll
        for (int r = 0; r < 8; ++r) {
            long long m = m0 + hi * 8 + r;
            float v = c[r] + bv;
            h[m * HID + gcol] = v;
            skip[m * HID + gcol] = v;
        }
    }
}

// ---------------------------------------------------------------------------
// Edge message + scatter:  agg[dst] += relu(h[src] + edge_attr@W_edge + b_edge)
// One edge per wave; e recomputed on the fly (saves ~700 MB/conv of HBM reads).
// ---------------------------------------------------------------------------
#define EDGES_PER_WAVE 8
__global__ void __launch_bounds__(256)
edge_kernel(const float* __restrict__ h, float* __restrict__ agg,
            const float* __restrict__ edge_attr,
            const int* __restrict__ src, const int* __restrict__ dst,
            const float* __restrict__ W_edge, const float* __restrict__ b_edge,
            long long nEdges)
{
    __shared__ float We[IN_EDGE * HID];   // 8 KB
    __shared__ float bE[HID];
    const int tid = threadIdx.x;
    for (int i = tid; i < IN_EDGE * HID; i += 256) We[i] = W_edge[i];
    if (tid < HID) bE[tid] = b_edge[tid];
    __syncthreads();

    const int lane = tid & 31;
    const int wave = tid >> 5;
    long long base = ((long long)blockIdx.x * 8 + wave) * EDGES_PER_WAVE;
    for (int t = 0; t < EDGES_PER_WAVE; ++t) {
        long long eid = base + t;
        if (eid >= nEdges) return;
        const int s = src[eid];
        const int d = dst[eid];
        float eav = (lane < IN_EDGE) ? edge_attr[eid * IN_EDGE + lane] : 0.f;
        float a0 = bE[lane], a1 = bE[lane + 32], a2 = bE[lane + 64], a3 = bE[lane + 96];
#pragma unroll
        for (int k = 0; k < IN_EDGE; ++k) {
            float ak = __shfl(eav, k, 32);
            a0 = fmaf(ak, We[k * HID + lane      ], a0);
            a1 = fmaf(ak, We[k * HID + lane + 32 ], a1);
            a2 = fmaf(ak, We[k * HID + lane + 64 ], a2);
            a3 = fmaf(ak, We[k * HID + lane + 96 ], a3);
        }
        const float* hs = h + (long long)s * HID;
        float m0 = hs[lane      ] + a0;
        float m1 = hs[lane + 32 ] + a1;
        float m2 = hs[lane + 64 ] + a2;
        float m3 = hs[lane + 96 ] + a3;
        float* ad = agg + (long long)d * HID;
        unsafeAtomicAdd(ad + lane,      m0 > 0.f ? m0 : 0.f);
        unsafeAtomicAdd(ad + lane + 32, m1 > 0.f ? m1 : 0.f);
        unsafeAtomicAdd(ad + lane + 64, m2 > 0.f ? m2 : 0.f);
        unsafeAtomicAdd(ad + lane + 96, m3 > 0.f ? m3 : 0.f);
    }
}

// ---------------------------------------------------------------------------
// h = leaky_relu((h + agg) @ W_conv + b_conv)   (K = 128, in-place on h)
// ---------------------------------------------------------------------------
__global__ void __launch_bounds__(256)
conv_kernel(float* __restrict__ h, const float* __restrict__ agg,
            const __bf16* __restrict__ Wc, const float* __restrict__ bias,
            int nTiles)
{
    constexpr int K = HID;
    __shared__ __align__(16) __bf16 wT[HID * K];   // 32 KB
    __shared__ __align__(16) __bf16 aT[16 * K];    // 4 KB
    __shared__ float bS[HID];

    const int tid = threadIdx.x;
    for (int i = tid; i < K * HID; i += 256) {
        int k = i >> 7, col = i & 127;
        wT[col * K + k] = Wc[i];
    }
    if (tid < HID) bS[tid] = bias[tid];

    const int lane = tid & 31, wave = tid >> 5;
    const int tile0 = blockIdx.x * TILES_PER_BLOCK;
    for (int t = 0; t < TILES_PER_BLOCK; ++t) {
        int tile = tile0 + t;
        if (tile >= nTiles) break;
        long long m0 = (long long)tile * 16;
        __syncthreads();
        for (int i = tid; i < 16 * K; i += 256) {
            int r = i >> 7, k = i & 127;
            long long idx = (m0 + r) * HID + k;
            aT[i] = (__bf16)(h[idx] + agg[idx]);
        }
        __syncthreads();
        v8f c = wmma_tile<K>(aT, wT, lane, wave);
        const int row = lane & 15, hi = lane >> 4;
        const int gcol = wave * 16 + row;
        const float bv = bS[gcol];
#pragma unroll
        for (int r = 0; r < 8; ++r) {
            long long m = m0 + hi * 8 + r;
            float v = c[r] + bv;
            h[m * HID + gcol] = (v >= 0.f) ? v : NEG_SLOPE * v;
        }
    }
}

// ---------------------------------------------------------------------------
// y = (skip + h) @ W_head + b_head ; LayerNorm(y) -> out
// ---------------------------------------------------------------------------
__global__ void __launch_bounds__(256)
head_kernel(const float* __restrict__ h, const float* __restrict__ skip,
            const __bf16* __restrict__ Wh, const float* __restrict__ bias,
            const float* __restrict__ gamma, const float* __restrict__ beta,
            float* __restrict__ out, int nTiles)
{
    constexpr int K = HID;
    __shared__ __align__(16) __bf16 wT[HID * K];   // 32 KB
    __shared__ __align__(16) __bf16 aT[16 * K];    // 4 KB
    __shared__ float yT[16 * OUT_DIM];             // 8 KB
    __shared__ float bS[OUT_DIM], gS[OUT_DIM], btS[OUT_DIM];
    __shared__ float muS[16], rsS[16];

    const int tid = threadIdx.x;
    for (int i = tid; i < K * HID; i += 256) {
        int k = i >> 7, col = i & 127;
        wT[col * K + k] = Wh[i];
    }
    if (tid < OUT_DIM) { bS[tid] = bias[tid]; gS[tid] = gamma[tid]; btS[tid] = beta[tid]; }

    const int lane = tid & 31, wave = tid >> 5;
    const int tile0 = blockIdx.x * TILES_PER_BLOCK;
    for (int t = 0; t < TILES_PER_BLOCK; ++t) {
        int tile = tile0 + t;
        if (tile >= nTiles) break;
        long long m0 = (long long)tile * 16;
        __syncthreads();
        for (int i = tid; i < 16 * K; i += 256) {
            int r = i >> 7, k = i & 127;
            long long idx = (m0 + r) * HID + k;
            aT[i] = (__bf16)(skip[idx] + h[idx]);
        }
        __syncthreads();
        v8f c = wmma_tile<K>(aT, wT, lane, wave);
        const int row = lane & 15, hi = lane >> 4;
        const int gcol = wave * 16 + row;
        const float bv = bS[gcol];
#pragma unroll
        for (int r = 0; r < 8; ++r)
            yT[(hi * 8 + r) * OUT_DIM + gcol] = c[r] + bv;
        __syncthreads();
        if (tid < 16) {
            float s = 0.f;
            for (int j = 0; j < OUT_DIM; ++j) s += yT[tid * OUT_DIM + j];
            float mu = s * (1.0f / OUT_DIM);
            float v = 0.f;
            for (int j = 0; j < OUT_DIM; ++j) {
                float dd = yT[tid * OUT_DIM + j] - mu;
                v = fmaf(dd, dd, v);
            }
            muS[tid] = mu;
            rsS[tid] = rsqrtf(v * (1.0f / OUT_DIM) + LN_EPS);
        }
        __syncthreads();
        for (int i = tid; i < 16 * OUT_DIM; i += 256) {
            int r = i >> 7, j = i & 127;
            out[(m0 + r) * OUT_DIM + j] =
                (yT[i] - muS[r]) * rsS[r] * gS[j] + btS[j];
        }
        __syncthreads();   // protect yT/aT before next tile
    }
}

// ---------------------------------------------------------------------------
extern "C" void kernel_launch(void* const* d_in, const int* in_sizes, int n_in,
                              void* d_out, int out_size, void* d_ws, size_t ws_size,
                              hipStream_t stream)
{
    const float* x       = (const float*)d_in[0];
    const float* ea      = (const float*)d_in[1];
    const int*   ei      = (const int*)  d_in[2];
    const float* W_node  = (const float*)d_in[3];
    const float* b_node  = (const float*)d_in[4];
    const float* W_edge  = (const float*)d_in[5];
    const float* b_edge  = (const float*)d_in[6];
    const float* W_conv  = (const float*)d_in[7];
    const float* b_conv  = (const float*)d_in[8];
    const float* W_head  = (const float*)d_in[9];
    const float* b_head  = (const float*)d_in[10];
    const float* ln_g    = (const float*)d_in[11];
    const float* ln_b    = (const float*)d_in[12];

    const long long nE = (long long)in_sizes[1] / IN_EDGE;
    const int* src = ei;
    const int* dst = ei + nE;

    // workspace layout
    float*  h    = (float*)d_ws;
    float*  skip = h    + (size_t)NN * HID;
    float*  agg  = skip + (size_t)NN * HID;
    __bf16* Wn   = (__bf16*)(agg + (size_t)NN * HID);
    __bf16* Wc   = Wn + IN_NODE * HID;
    __bf16* Wh   = Wc + N_CONVS * HID * HID;

    // weight conversion f32 -> bf16
    {
        int n1 = IN_NODE * HID, n2 = N_CONVS * HID * HID, n3 = HID * OUT_DIM;
        cvt_bf16_kernel<<<(n1 + 255) / 256, 256, 0, stream>>>(W_node, Wn, n1);
        cvt_bf16_kernel<<<(n2 + 255) / 256, 256, 0, stream>>>(W_conv, Wc, n2);
        cvt_bf16_kernel<<<(n3 + 255) / 256, 256, 0, stream>>>(W_head, Wh, n3);
    }

    const int nTiles = NN / 16;                               // 6250
    const int gemmGrid = (nTiles + TILES_PER_BLOCK - 1) / TILES_PER_BLOCK;

    node_kernel<<<gemmGrid, 256, 0, stream>>>(x, Wn, b_node, h, skip, nTiles);

    const long long n4 = (long long)NN * HID / 4;
    const int zeroGrid = (int)((n4 + 255) / 256);
    const int edgeGrid = (int)((nE + 63) / 64);               // 64 edges / block

    for (int l = 0; l < N_CONVS; ++l) {
        zero_kernel<<<zeroGrid, 256, 0, stream>>>(agg, n4);
        edge_kernel<<<edgeGrid, 256, 0, stream>>>(h, agg, ea, src, dst,
                                                  W_edge, b_edge, nE);
        conv_kernel<<<gemmGrid, 256, 0, stream>>>(h, agg,
                                                  Wc + (size_t)l * HID * HID,
                                                  b_conv + (size_t)l * HID, nTiles);
    }

    head_kernel<<<gemmGrid, 256, 0, stream>>>(h, skip, Wh, b_head,
                                              ln_g, ln_b, (float*)d_out, nTiles);
}
